// ScaleDotProductAttention_85899346375
// MI455X (gfx1250) — matrix operand order repaired and checked by reading, not compiled, not run on hardware
//
#include <hip/hip_runtime.h>
#include <hip/hip_bf16.h>

typedef _Float16 v16h __attribute__((ext_vector_type(16)));
typedef _Float16 h2   __attribute__((ext_vector_type(2)));
typedef float    v8f  __attribute__((ext_vector_type(8)));

#define SLEN  2048
#define DDIM  128
#define WPB   2   // waves per block

// v_cvt_pk_rtz_f16_f32: pack two f32 into two f16 in one dword.
// Builtin returns __fp16x2 -> launder to _Float16x2.
static __device__ inline h2 pkrtz(float a, float b) {
    return __builtin_bit_cast(h2, __builtin_amdgcn_cvt_pkrtz(a, b));
}

__global__ __launch_bounds__(WPB * 32)
void sdpa_fused_kernel(const float* __restrict__ qg,
                       const float* __restrict__ kg,
                       const float* __restrict__ vg,
                       float* __restrict__ outg,
                       float* __restrict__ scoreg)
{
    // Per-wave LDS (no cross-wave sharing -> no block barriers needed)
    __shared__ _Float16 sKT[WPB][DDIM * 32];  // K block TRANSPOSED: [d 0..127][key 0..31] f16
    __shared__ _Float16 sV [WPB][32 * DDIM];  // V block row-major : [key][d] f16
    __shared__ _Float16 sPA[WPB][32 * 16];    // P tile in per-lane A-operand layout: [lane][h]

    const int wv   = threadIdx.x >> 5;
    const int lane = threadIdx.x & 31;
    const int hv   = lane >> 4;     // lane half (0/1)
    const int mrow = lane & 15;     // M row (A) / N col (C,D)
    const int lo   = hv * 8;        // A-operand K-interleave offset

    const int gw = blockIdx.x * WPB + wv;   // 0..4095 global wave id
    const int bh = gw >> 7;                 // head (B*H = 32)
    const int qt = gw & 127;                // query tile (S/16 = 128)
    const int q0 = qt * 16;

    const float* qb = qg + (size_t)bh * SLEN * DDIM;
    const float* kb = kg + (size_t)bh * SLEN * DDIM;
    const float* vb = vg + (size_t)bh * SLEN * DDIM;
    float* ob = outg   + (size_t)bh * SLEN * DDIM;
    float* sb = scoreg + (size_t)bh * SLEN * SLEN;

    _Float16* smKT = sKT[wv];
    _Float16* smV  = sV [wv];
    _Float16* smPA = sPA[wv];

    // ---- Q tile -> A-operand registers (4 chunks of K=32), f32 -> f16 ----
    // A 16x32 f16: lanes<16: h0..7=K0..7, h8..15=K16..23 ; lanes>=16: +8
    v16h aq[4];
    {
        const float* qr = qb + (size_t)(q0 + mrow) * DDIM;
        #pragma unroll
        for (int c = 0; c < 4; ++c) {
            #pragma unroll
            for (int j = 0; j < 8; ++j) {
                aq[c][j]     = (_Float16)qr[c * 32 + lo + j];
                aq[c][j + 8] = (_Float16)qr[c * 32 + 16 + lo + j];
            }
        }
    }

    // ---- stage K block (32 keys x 128 d) transposed into LDS as f16 ----
    // Lane kp = lane&15 owns key-pair (2kp, 2kp+1); ph selects 16B column half.
    // Each dword of smKT holds keys (2kp,2kp+1) at one d -> v_cvt_pk_rtz packs.
    // Per-store banks = j*16 + kp over 16 kp values -> ~2-way conflicts.
    auto stageKT = [&](const float* src) {
        const int kp = lane & 15;
        const int ph = lane >> 4;
        const float* rA = src + (size_t)(2 * kp) * DDIM;
        const float* rB = rA + DDIM;
        #pragma unroll
        for (int dg = 0; dg < 16; ++dg) {
            const int d0 = dg * 8 + ph * 4;
            float4 a = *(const float4*)(rA + d0);
            float4 b = *(const float4*)(rB + d0);
            *(h2*)(smKT + (d0 + 0) * 32 + kp * 2) = pkrtz(a.x, b.x);
            *(h2*)(smKT + (d0 + 1) * 32 + kp * 2) = pkrtz(a.y, b.y);
            *(h2*)(smKT + (d0 + 2) * 32 + kp * 2) = pkrtz(a.z, b.z);
            *(h2*)(smKT + (d0 + 3) * 32 + kp * 2) = pkrtz(a.w, b.w);
        }
    };

    // ---- stage V block (32 keys x 128 d) row-major into LDS as f16 ----
    auto stageV = [&](const float* src) {
        const float4* s4 = (const float4*)src;   // 1024 float4 per block
        #pragma unroll
        for (int it = 0; it < 32; ++it) {
            int idx = it * 32 + lane;
            float4 f = s4[idx];
            uint2 w;
            w.x = __builtin_bit_cast(unsigned int, pkrtz(f.x, f.y));
            w.y = __builtin_bit_cast(unsigned int, pkrtz(f.z, f.w));
            *(uint2*)(smV + idx * 4) = w;        // ds_store_b64
        }
    };

    // ---- S subtile (16 q x 16 keys), contraction over D=128 ----
    // B 32x16 f16: lane%16 = key col, elements = d in table K order (same as A),
    // gathered per element from transposed smKT.
    auto computeS = [&](int sub) -> v8f {
        v8f acc = {};
        #pragma unroll
        for (int c = 0; c < 4; ++c) {
            v16h bkv;
            #pragma unroll
            for (int s = 0; s < 16; ++s)
                bkv[s] = smKT[(c * 32 + lo + (s & 7) + (s >> 3) * 16) * 32 + sub * 16 + mrow];
            acc = __builtin_amdgcn_wmma_f32_16x16x32_f16(
                      false, aq[c], false, bkv, (short)0, acc, false, false);
        }
        return acc;
    };

    const float scale = 0.08838834764831845f;   // 1/sqrt(128)
    const int   kbmax = qt >> 1;                // last 32-key block in causal region

    // ================= Pass 1: online row max / sum(exp) =================
    float m[8], l[8];
    #pragma unroll
    for (int r = 0; r < 8; ++r) { m[r] = -1e30f; l[r] = 0.0f; }

    for (int kblk = 0; kblk <= kbmax; ++kblk) {
        if (kblk < kbmax)
            __builtin_prefetch(kb + (size_t)(kblk + 1) * 32 * DDIM + lane * 16, 0, 1);
        stageKT(kb + (size_t)kblk * 32 * DDIM);
        asm volatile("s_wait_dscnt 0" ::: "memory");
        #pragma unroll
        for (int sub = 0; sub < 2; ++sub) {
            v8f s = computeS(sub);
            int kcol = kblk * 32 + sub * 16 + mrow;
            #pragma unroll
            for (int r = 0; r < 8; ++r) {
                int qrow = q0 + r + hv * 8;
                float sv = (kcol <= qrow) ? s[r] * scale : -10000.0f;
                float tmax = sv;
                tmax = fmaxf(tmax, __shfl_xor(tmax, 1));
                tmax = fmaxf(tmax, __shfl_xor(tmax, 2));
                tmax = fmaxf(tmax, __shfl_xor(tmax, 4));
                tmax = fmaxf(tmax, __shfl_xor(tmax, 8));
                float mn = fmaxf(m[r], tmax);
                float pe = __expf(sv - mn);
                float ps = pe;
                ps += __shfl_xor(ps, 1);
                ps += __shfl_xor(ps, 2);
                ps += __shfl_xor(ps, 4);
                ps += __shfl_xor(ps, 8);
                l[r] = l[r] * __expf(m[r] - mn) + ps;
                m[r] = mn;
            }
        }
    }

    float linv[8];
    #pragma unroll
    for (int r = 0; r < 8; ++r) linv[r] = 1.0f / l[r];

    // ====== Pass 2: recompute S, write normalized score, O += P*V ======
    v8f o[8];
    #pragma unroll
    for (int n = 0; n < 8; ++n) o[n] = (v8f){};

    for (int kblk = 0; kblk < SLEN / 32; ++kblk) {
        if (kblk <= kbmax) {
            if (kblk < kbmax)
                __builtin_prefetch(kb + (size_t)(kblk + 1) * 32 * DDIM + lane * 16, 0, 1);
            stageKT(kb + (size_t)kblk * 32 * DDIM);
            stageV (vb + (size_t)kblk * 32 * DDIM);
            asm volatile("s_wait_dscnt 0" ::: "memory");

            #pragma unroll
            for (int sub = 0; sub < 2; ++sub) {
                v8f s = computeS(sub);
                int kcol = kblk * 32 + sub * 16 + mrow;
                #pragma unroll
                for (int r = 0; r < 8; ++r) {
                    int qrow = q0 + r + hv * 8;
                    float sv = (kcol <= qrow) ? s[r] * scale : -10000.0f;
                    float p = __expf(sv - m[r]) * linv[r];   // masked -> exactly 0.0
                    sb[(size_t)qrow * SLEN + kcol] = p;
                    // scatter into per-lane A-operand layout:
                    // dest lane = (mrow>=8 ? 16:0) + qrow_local ; half = (mrow&7)+sub*8
                    smPA[(((mrow >> 3) << 4) + r + hv * 8) * 16 + (mrow & 7) + sub * 8]
                        = (_Float16)p;
                }
            }
            asm volatile("s_wait_dscnt 0" ::: "memory");

            // A operand: P 16x32 f16 -> contiguous 32B per lane
            v16h ap = *(const v16h*)(smPA + lane * 16);
            // O[16x128] += P[16x32] * V[32x128], 8 column chunks of 16
            #pragma unroll
            for (int n = 0; n < 8; ++n) {
                v16h bv;
                #pragma unroll
                for (int s = 0; s < 16; ++s)
                    bv[s] = smV[(lo + (s & 7) + (s >> 3) * 16) * DDIM + n * 16 + mrow];
                o[n] = __builtin_amdgcn_wmma_f32_16x16x32_f16(
                           false, ap, false, bv, (short)0, o[n], false, false);
            }
        } else {
            // fully masked upper-triangle region: softmax underflows to 0.0
            const float4 z4 = make_float4(0.f, 0.f, 0.f, 0.f);
            #pragma unroll
            for (int it = 0; it < 4; ++it) {
                int idx = it * 32 + lane;            // 128 float4 = 16 rows x 32 cols
                int row = idx >> 3, c4 = idx & 7;
                *(float4*)(sb + (size_t)(q0 + row) * SLEN + kblk * 32 + c4 * 4) = z4;
            }
        }
    }

    // ---- write out tile: C/D layout -> global (coalesced 64B segments) ----
    #pragma unroll
    for (int n = 0; n < 8; ++n)
        #pragma unroll
        for (int r = 0; r < 8; ++r)
            ob[(size_t)(q0 + r + hv * 8) * DDIM + n * 16 + mrow] = o[n][r];
}

extern "C" void kernel_launch(void* const* d_in, const int* in_sizes, int n_in,
                              void* d_out, int out_size, void* d_ws, size_t ws_size,
                              hipStream_t stream) {
    const float* q = (const float*)d_in[0];
    const float* k = (const float*)d_in[1];
    const float* v = (const float*)d_in[2];
    // d_in[3]: causal mask (int32 tril) — handled analytically in-kernel.
    float* outp   = (float*)d_out;
    float* scorep = outp + (size_t)2 * 16 * SLEN * DDIM;   // tuple: (out, score)

    const int total_waves = 2 * 16 * (SLEN / 16);          // 4096
    dim3 grid(total_waves / WPB);
    dim3 block(WPB * 32);
    sdpa_fused_kernel<<<grid, block, 0, stream>>>(q, k, v, outp, scorep);
}
